// ViTSlotCompetition_4707284156559
// MI455X (gfx1250) — matrix operand order repaired and checked by reading, not compiled, not run on hardware
//
#include <hip/hip_runtime.h>
#include <hip/hip_bf16.h>
#include <math.h>

// ---------------------------------------------------------------------------
// Types & helpers
// ---------------------------------------------------------------------------
typedef unsigned short u16;
typedef __attribute__((ext_vector_type(16))) __bf16 v16bf;
typedef __attribute__((ext_vector_type(8)))  float  v8f;

union Frag16 {
    v16bf v;
    u16   u[16];
    uint4 q[2];
};

__device__ __forceinline__ u16 f32_to_bf16(float f) {
    unsigned int x = __float_as_uint(f);
    unsigned int r = (x + 0x7FFFu + ((x >> 16) & 1u)) >> 16;   // round-nearest-even
    return (u16)r;
}
__device__ __forceinline__ float bf16_to_f32(u16 u) {
    return __uint_as_float(((unsigned int)u) << 16);
}

static const int Dd  = 768;
static const int Hh  = 3072;
static const int Bb  = 128;
static const int Pp  = 196;
static const int NPB = 392;          // S*P patches per batch
static const int BS  = 256;          // B*S slot rows
static const int NROW = Bb * NPB;    // 50176 patch rows

// LDS tile geometry for the WMMA GEMM
static const int LDS_STRIDE = 72;    // 64 data + 8 pad (row = 144 B, 16B aligned)

// ---------------------------------------------------------------------------
// 1) slots init: mean over P axis of [B,S,P,D]
// ---------------------------------------------------------------------------
__global__ __launch_bounds__(256) void slots_init_kernel(
    const float* __restrict__ pf, float* __restrict__ slots) {
    int row = blockIdx.x;                 // b*2+s
    int t   = threadIdx.x;
    const float* base = pf + (size_t)row * Pp * Dd;
    #pragma unroll
    for (int e = 0; e < 3; ++e) {
        int d = t + 256 * e;
        float s = 0.f;
        for (int p = 0; p < Pp; ++p) s += base[(size_t)p * Dd + d];
        slots[(size_t)row * Dd + d] = s * (1.0f / 196.0f);
    }
}

// ---------------------------------------------------------------------------
// 2) row LayerNorm (f32 in, bf16 out), rows of length 768
// ---------------------------------------------------------------------------
__global__ __launch_bounds__(256) void ln_rows_kernel(
    const float* __restrict__ x, const float* __restrict__ g,
    const float* __restrict__ be, u16* __restrict__ out) {
    int row = blockIdx.x, t = threadIdx.x;
    int lane = t & 31, wid = t >> 5;
    const float* xr = x + (size_t)row * Dd;
    float v[3], s = 0.f, ss = 0.f;
    #pragma unroll
    for (int e = 0; e < 3; ++e) {
        v[e] = xr[t + 256 * e];
        s += v[e]; ss += v[e] * v[e];
    }
    #pragma unroll
    for (int o = 16; o > 0; o >>= 1) {
        s  += __shfl_down(s, o);
        ss += __shfl_down(ss, o);
    }
    __shared__ float rs[8], rss[8], sh_mu, sh_rstd;
    if (lane == 0) { rs[wid] = s; rss[wid] = ss; }
    __syncthreads();
    if (t == 0) {
        float S = 0.f, SS = 0.f;
        #pragma unroll
        for (int i = 0; i < 8; ++i) { S += rs[i]; SS += rss[i]; }
        float mu  = S * (1.0f / 768.0f);
        float var = SS * (1.0f / 768.0f) - mu * mu;
        sh_mu = mu; sh_rstd = rsqrtf(var + 1e-5f);
    }
    __syncthreads();
    float mu = sh_mu, rstd = sh_rstd;
    #pragma unroll
    for (int e = 0; e < 3; ++e) {
        int d = t + 256 * e;
        out[(size_t)row * Dd + d] = f32_to_bf16((v[e] - mu) * rstd * g[d] + be[d]);
    }
}

// ---------------------------------------------------------------------------
// 3) cast f32 -> bf16
// ---------------------------------------------------------------------------
__global__ __launch_bounds__(256) void cast_bf16_kernel(
    const float* __restrict__ in, u16* __restrict__ out, int n) {
    int i = blockIdx.x * 256 + threadIdx.x;
    if (i < n) out[i] = f32_to_bf16(in[i]);
}

// ---------------------------------------------------------------------------
// 4) GEMM  C[m,n] = alpha * sum_k A[m,k]*W[n,k] + bias[n] (+C)
//    A: bf16 [M,K] row-major, W: bf16 [N,K] row-major.
//    Block (8 waves) computes 256x64; each wave owns a 32x64 tile (two
//    16-row subtiles sharing the B fragments -> 16 WMMA per 24 ds_load_b128).
//    K staged in 64-wide chunks through LDS with double-buffered
//    global_load_async_to_lds_b128 (ASYNCcnt pipeline).
//    Requires M%256==0, N%64==0, K%64==0 (true for all GEMMs here).
// ---------------------------------------------------------------------------
__global__ __launch_bounds__(256) void gemm_nt_wmma_kernel(
    const u16* __restrict__ A, int lda,
    const u16* __restrict__ W, int ldw,
    const float* __restrict__ bias,
    float* __restrict__ Cf, u16* __restrict__ Cb, int ldc,
    int M, int N, int K, float alpha, int relu, int addC) {
    __shared__ __align__(16) u16 As[2][256 * LDS_STRIDE];
    __shared__ __align__(16) u16 Bs[2][64 * LDS_STRIDE];

    int t    = threadIdx.x;
    int wv   = t >> 5;
    int lane = t & 31;
    int l15  = lane & 15, lhi = lane >> 4;

    int Mb = M >> 8;
    int mB = (int)blockIdx.x % Mb;
    int nB = (int)blockIdx.x / Mb;

    const u16* Ablk = A + ((size_t)mB << 8) * lda;
    const u16* Wblk = W + ((size_t)nB << 6) * ldw;

    unsigned asb[2] = { (unsigned)(uintptr_t)&As[0][0],
                        (unsigned)(uintptr_t)&As[1][0] };
    unsigned bsb[2] = { (unsigned)(uintptr_t)&Bs[0][0],
                        (unsigned)(uintptr_t)&Bs[1][0] };

    v8f acc[2][4];
    #pragma unroll
    for (int h = 0; h < 2; ++h)
        #pragma unroll
        for (int j = 0; j < 4; ++j)
            #pragma unroll
            for (int r = 0; r < 8; ++r) acc[h][j][r] = 0.f;

    // issue one chunk's async copies (10 x b128 per thread -> ASYNCcnt += 10/wave)
    auto issue_chunk = [&](int k0, int p) {
        #pragma unroll
        for (int i = 0; i < 8; ++i) {                 // A: 256 rows x 8 segs
            int idx = t + (i << 8);
            int row = idx >> 3, seg = idx & 7;
            const u16* g = Ablk + (size_t)row * lda + k0 + (seg << 3);
            unsigned l = asb[p] + (unsigned)(row * (LDS_STRIDE * 2) + (seg << 4));
            asm volatile("global_load_async_to_lds_b128 %0, %1, off"
                         :: "v"(l), "v"(g) : "memory");
        }
        #pragma unroll
        for (int i = 0; i < 2; ++i) {                 // B: 64 rows x 8 segs
            int idx = t + (i << 8);
            int row = idx >> 3, seg = idx & 7;
            const u16* g = Wblk + (size_t)row * ldw + k0 + (seg << 3);
            unsigned l = bsb[p] + (unsigned)(row * (LDS_STRIDE * 2) + (seg << 4));
            asm volatile("global_load_async_to_lds_b128 %0, %1, off"
                         :: "v"(l), "v"(g) : "memory");
        }
    };

    int nCh = K >> 6;
    issue_chunk(0, 0);
    for (int c = 0; c < nCh; ++c) {
        int p = c & 1;
        if (c + 1 < nCh) {
            issue_chunk((c + 1) << 6, p ^ 1);
            asm volatile("s_wait_asynccnt 0xa" ::: "memory");  // own chunk-c done
        } else {
            asm volatile("s_wait_asynccnt 0x0" ::: "memory");
        }
        __syncthreads();                                       // all waves' parts done

        #pragma unroll
        for (int kk = 0; kk < 64; kk += 32) {
            Frag16 a0, a1, bfr[4];
            const u16* ar0 = &As[p][(wv * 32 + l15) * LDS_STRIDE + kk + (lhi << 3)];
            const u16* ar1 = &As[p][(wv * 32 + 16 + l15) * LDS_STRIDE + kk + (lhi << 3)];
            a0.q[0] = *(const uint4*)(ar0);           // K = kk+kh .. +7
            a0.q[1] = *(const uint4*)(ar0 + 16);      // K = kk+16+kh .. +7
            a1.q[0] = *(const uint4*)(ar1);
            a1.q[1] = *(const uint4*)(ar1 + 16);
            #pragma unroll
            for (int j = 0; j < 4; ++j) {
                const u16* brow = &Bs[p][(j * 16 + l15) * LDS_STRIDE + kk + (lhi << 3)];
                bfr[j].q[0] = *(const uint4*)(brow);
                bfr[j].q[1] = *(const uint4*)(brow + 16);
            }
            #pragma unroll
            for (int j = 0; j < 4; ++j) {
                acc[0][j] = __builtin_amdgcn_wmma_f32_16x16x32_bf16(
                    false, a0.v, false, bfr[j].v, (short)0, acc[0][j], false, false);
                acc[1][j] = __builtin_amdgcn_wmma_f32_16x16x32_bf16(
                    false, a1.v, false, bfr[j].v, (short)0, acc[1][j], false, false);
            }
        }
        __syncthreads();   // protect buffer p before it is refilled next iter
    }

    #pragma unroll
    for (int h = 0; h < 2; ++h) {
        int row0 = (mB << 8) + wv * 32 + h * 16 + (lhi << 3);   // C/D layout rows
        #pragma unroll
        for (int j = 0; j < 4; ++j) {
            int col = (nB << 6) + j * 16 + l15;
            float bb = bias ? bias[col] : 0.f;
            #pragma unroll
            for (int r = 0; r < 8; ++r) {
                float vv = alpha * acc[h][j][r] + bb;
                if (relu) vv = vv > 0.f ? vv : 0.f;
                size_t idx = (size_t)(row0 + r) * ldc + col;
                if (addC) vv += Cf[idx];
                if (Cf) Cf[idx] = vv;
                if (Cb) Cb[idx] = f32_to_bf16(vv);
            }
        }
    }
}

// ---------------------------------------------------------------------------
// 5) attention logits + softmax(S=2) + EPS : one wave per patch row
// ---------------------------------------------------------------------------
__global__ __launch_bounds__(256) void attn_kernel(
    const u16* __restrict__ k, const float* __restrict__ q,
    float* __restrict__ attn) {
    int wid = threadIdx.x >> 5, lane = threadIdx.x & 31;
    int r = blockIdx.x * 8 + wid;          // 0..NROW-1
    int b = r / NPB;
    const u16*   krow = k + (size_t)r * Dd;
    const float* q0   = q + (size_t)(b * 2) * Dd;
    const float* q1   = q0 + Dd;
    float d0 = 0.f, d1 = 0.f;
    for (int i = lane; i < Dd; i += 32) {
        float kk = bf16_to_f32(krow[i]);
        d0 += kk * q0[i];
        d1 += kk * q1[i];
    }
    #pragma unroll
    for (int o = 16; o > 0; o >>= 1) {
        d0 += __shfl_down(d0, o);
        d1 += __shfl_down(d1, o);
    }
    if (lane == 0) {
        float m  = fmaxf(d0, d1);
        float e0 = __expf(d0 - m), e1 = __expf(d1 - m);
        float inv = 1.f / (e0 + e1);
        attn[(size_t)r * 2 + 0] = e0 * inv + 1e-8f;
        attn[(size_t)r * 2 + 1] = e1 * inv + 1e-8f;
    }
}

// ---------------------------------------------------------------------------
// 6) updates = (attn / (colsum+EPS))^T v  per batch; deterministic colsum
// ---------------------------------------------------------------------------
__global__ __launch_bounds__(256) void updates_kernel(
    const float* __restrict__ attn, const u16* __restrict__ v,
    u16* __restrict__ upd) {
    int b = blockIdx.x, t = threadIdx.x;
    __shared__ float s_attn[256];   // 128 n * 2 slots
    float acc[2][3];
    #pragma unroll
    for (int s = 0; s < 2; ++s)
        #pragma unroll
        for (int e = 0; e < 3; ++e) acc[s][e] = 0.f;
    float c0 = 0.f, c1 = 0.f;

    for (int n0 = 0; n0 < NPB; n0 += 128) {
        int cs = NPB - n0; if (cs > 128) cs = 128;
        __syncthreads();
        if (t < cs * 2) s_attn[t] = attn[(size_t)(b * NPB + n0) * 2 + t];
        __syncthreads();
        for (int i = 0; i < cs; ++i) {
            float a0 = s_attn[2 * i], a1 = s_attn[2 * i + 1];
            c0 += a0; c1 += a1;
            const u16* vrow = v + (size_t)(b * NPB + n0 + i) * Dd;
            #pragma unroll
            for (int e = 0; e < 3; ++e) {
                float vv = bf16_to_f32(vrow[t + 256 * e]);
                acc[0][e] += a0 * vv;
                acc[1][e] += a1 * vv;
            }
        }
    }
    float inv0 = 1.f / (c0 + 1e-8f), inv1 = 1.f / (c1 + 1e-8f);
    #pragma unroll
    for (int e = 0; e < 3; ++e) {
        int d = t + 256 * e;
        upd[(size_t)(b * 2 + 0) * Dd + d] = f32_to_bf16(acc[0][e] * inv0);
        upd[(size_t)(b * 2 + 1) * Dd + d] = f32_to_bf16(acc[1][e] * inv1);
    }
}

// ---------------------------------------------------------------------------
// 7) GRU elementwise combine (gates order r,z,n; biases folded into GEMMs)
// ---------------------------------------------------------------------------
__global__ __launch_bounds__(256) void gru_combine_kernel(
    const float* __restrict__ gi, const float* __restrict__ gh,
    float* __restrict__ slots) {
    int idx = blockIdx.x * 256 + threadIdx.x;    // 0..BS*D-1
    int row = idx / Dd, d = idx % Dd;
    size_t g = (size_t)row * (3 * Dd) + d;
    float ir = gi[g],            hr = gh[g];
    float iz = gi[g + Dd],       hz = gh[g + Dd];
    float in_ = gi[g + 2 * Dd],  hn = gh[g + 2 * Dd];
    float r = 1.f / (1.f + __expf(-(ir + hr)));
    float z = 1.f / (1.f + __expf(-(iz + hz)));
    float n = tanhf(in_ + r * hn);
    float h = slots[idx];
    slots[idx] = (1.f - z) * n + z * h;
}

// ---------------------------------------------------------------------------
// Host orchestration
// ---------------------------------------------------------------------------
extern "C" void kernel_launch(void* const* d_in, const int* in_sizes, int n_in,
                              void* d_out, int out_size, void* d_ws, size_t ws_size,
                              hipStream_t stream) {
    const float* pf    = (const float*)d_in[0];
    const float* ns_g  = (const float*)d_in[1];
    const float* ns_b  = (const float*)d_in[2];
    const float* ni_g  = (const float*)d_in[3];
    const float* ni_b  = (const float*)d_in[4];
    const float* Wq    = (const float*)d_in[5];
    const float* Wk    = (const float*)d_in[6];
    const float* Wv    = (const float*)d_in[7];
    const float* W_ih  = (const float*)d_in[8];
    const float* W_hh  = (const float*)d_in[9];
    const float* b_ih  = (const float*)d_in[10];
    const float* b_hh  = (const float*)d_in[11];
    const float* mln_g = (const float*)d_in[12];
    const float* mln_b = (const float*)d_in[13];
    const float* W1    = (const float*)d_in[14];
    const float* b1    = (const float*)d_in[15];
    const float* W2    = (const float*)d_in[16];
    const float* b2    = (const float*)d_in[17];

    float* slots = (float*)d_out;                  // [256, 768] fp32

    // ---- workspace carve ----
    char* wsp = (char*)d_ws; size_t off = 0;
    auto carve = [&](size_t bytes) -> void* {
        void* p = wsp + off;
        off = (off + bytes + 255) & ~(size_t)255;
        return p;
    };
    u16* pn_b   = (u16*)carve((size_t)NROW * Dd * 2);
    u16* k_b    = (u16*)carve((size_t)NROW * Dd * 2);
    u16* v_b    = (u16*)carve((size_t)NROW * Dd * 2);
    u16* wq_b   = (u16*)carve((size_t)Dd * Dd * 2);
    u16* wk_b   = (u16*)carve((size_t)Dd * Dd * 2);
    u16* wv_b   = (u16*)carve((size_t)Dd * Dd * 2);
    u16* wih_b  = (u16*)carve((size_t)3 * Dd * Dd * 2);
    u16* whh_b  = (u16*)carve((size_t)3 * Dd * Dd * 2);
    u16* w1_b   = (u16*)carve((size_t)Hh * Dd * 2);
    u16* w2_b   = (u16*)carve((size_t)Dd * Hh * 2);
    u16* sn_b   = (u16*)carve((size_t)BS * Dd * 2);
    u16* sl_b   = (u16*)carve((size_t)BS * Dd * 2);
    u16* m_b    = (u16*)carve((size_t)BS * Dd * 2);
    u16* up_b   = (u16*)carve((size_t)BS * Dd * 2);
    u16* h1_b   = (u16*)carve((size_t)BS * Hh * 2);
    float* qbuf = (float*)carve((size_t)BS * Dd * 4);
    float* attn = (float*)carve((size_t)NROW * 2 * 4);
    float* gi   = (float*)carve((size_t)BS * 3 * Dd * 4);
    float* gh   = (float*)carve((size_t)BS * 3 * Dd * 4);
    (void)ws_size; (void)n_in; (void)in_sizes; (void)out_size;

    auto castN = [&](const float* src, u16* dst, int n) {
        cast_bf16_kernel<<<(n + 255) / 256, 256, 0, stream>>>(src, dst, n);
    };
    auto gemm = [&](const u16* A, int lda, const u16* W, int ldw,
                    const float* bias, float* Cf, u16* Cb, int ldc,
                    int M, int N, int K, float alpha, int relu, int addC) {
        int blocks = (M >> 8) * (N >> 6);
        gemm_nt_wmma_kernel<<<blocks, 256, 0, stream>>>(
            A, lda, W, ldw, bias, Cf, Cb, ldc, M, N, K, alpha, relu, addC);
    };

    // ---- weight casts (once per launch) ----
    castN(Wq, wq_b, Dd * Dd);
    castN(Wk, wk_b, Dd * Dd);
    castN(Wv, wv_b, Dd * Dd);
    castN(W_ih, wih_b, 3 * Dd * Dd);
    castN(W_hh, whh_b, 3 * Dd * Dd);
    castN(W1, w1_b, Hh * Dd);
    castN(W2, w2_b, Dd * Hh);

    // ---- slots = mean over patches; pn = LN(patches) ----
    slots_init_kernel<<<BS, 256, 0, stream>>>(pf, slots);
    ln_rows_kernel<<<NROW, 256, 0, stream>>>(pf, ni_g, ni_b, pn_b);

    // ---- k, v (big compute-bound GEMMs, bf16 out) ----
    gemm(pn_b, Dd, wk_b, Dd, nullptr, nullptr, k_b, Dd, NROW, Dd, Dd, 1.f, 0, 0);
    gemm(pn_b, Dd, wv_b, Dd, nullptr, nullptr, v_b, Dd, NROW, Dd, Dd, 1.f, 0, 0);

    const float scale = 1.0f / sqrtf(768.0f);

    for (int it = 0; it < 3; ++it) {
        // q = LN(slots) @ Wq^T * scale
        ln_rows_kernel<<<BS, 256, 0, stream>>>(slots, ns_g, ns_b, sn_b);
        gemm(sn_b, Dd, wq_b, Dd, nullptr, qbuf, nullptr, Dd, BS, Dd, Dd, scale, 0, 0);

        // attn + normalized updates
        attn_kernel<<<NROW / 8, 256, 0, stream>>>(k_b, qbuf, attn);
        updates_kernel<<<Bb, 256, 0, stream>>>(attn, v_b, up_b);

        // GRU: gi = upd @ W_ih^T + b_ih ; gh = slots @ W_hh^T + b_hh
        castN(slots, sl_b, BS * Dd);
        gemm(up_b, Dd, wih_b, Dd, b_ih, gi, nullptr, 3 * Dd, BS, 3 * Dd, Dd, 1.f, 0, 0);
        gemm(sl_b, Dd, whh_b, Dd, b_hh, gh, nullptr, 3 * Dd, BS, 3 * Dd, Dd, 1.f, 0, 0);
        gru_combine_kernel<<<(BS * Dd) / 256, 256, 0, stream>>>(gi, gh, slots);

        // MLP residual: slots += relu(LN(slots)@W1^T + b1) @ W2^T + b2
        ln_rows_kernel<<<BS, 256, 0, stream>>>(slots, mln_g, mln_b, m_b);
        gemm(m_b, Dd, w1_b, Dd, b1, nullptr, h1_b, Hh, BS, Hh, Dd, 1.f, 1, 0);
        gemm(h1_b, Hh, w2_b, Hh, b2, slots, nullptr, Dd, BS, Dd, Hh, 1.f, 0, 1);
    }
}
